// UpsampleConv2D_40604620817123
// MI455X (gfx1250) — matrix operand order repaired and checked by reading, not compiled, not run on hardware
//
#include <hip/hip_runtime.h>
#include <hip/hip_bf16.h>

typedef __attribute__((ext_vector_type(16))) __bf16 v16bf;
typedef __attribute__((ext_vector_type(8)))  __bf16 v8bf;
typedef __attribute__((ext_vector_type(8)))  float  v8f;
typedef __attribute__((ext_vector_type(4)))  unsigned int v4u;
typedef __attribute__((ext_vector_type(8)))  int    v8i;
typedef __attribute__((ext_vector_type(4)))  int    v4i;

#define N_      16
#define H_      64
#define W_      64
#define CIN     256
#define COUT    128
#define LAT_    512

// ---------------------------------------------------------------------------
// k1a: styles[n,i] = latents[n,:] @ style_w[:,i] + style_b[i] + 1
// ---------------------------------------------------------------------------
__global__ __launch_bounds__(256) void k_styles(const float* __restrict__ latents,
                                                const float* __restrict__ style_w,
                                                const float* __restrict__ style_b,
                                                float* __restrict__ styles) {
    int n = blockIdx.x;
    int i = threadIdx.x;                       // 0..255
    float acc = style_b[i] + 1.0f;
    const float* lat = latents + (size_t)n * LAT_;
    for (int l = 0; l < LAT_; ++l)
        acc += lat[l] * style_w[(size_t)l * CIN + i];
    styles[(size_t)n * CIN + i] = acc;
}

// ---------------------------------------------------------------------------
// k1b: wsq[o,i] = sum_{ky,kx} w[ky,kx,o,i]^2
// ---------------------------------------------------------------------------
__global__ __launch_bounds__(256) void k_wsq(const float* __restrict__ w,
                                             float* __restrict__ wsq) {
    int o = blockIdx.x;                        // 0..127
    int i = threadIdx.x;                       // 0..255
    float s = 0.f;
#pragma unroll
    for (int t = 0; t < 9; ++t) {
        float v = w[((size_t)t * COUT + o) * CIN + i];
        s += v * v;
    }
    wsq[(size_t)o * CIN + i] = s;
}

// ---------------------------------------------------------------------------
// k1c: d[n,o] = rsqrt( sum_i styles[n,i]^2 * wsq[o,i] + 1e-8 )
// ---------------------------------------------------------------------------
__global__ __launch_bounds__(128) void k_demod(const float* __restrict__ styles,
                                               const float* __restrict__ wsq,
                                               float* __restrict__ dmod) {
    int n = blockIdx.x;
    int o = threadIdx.x;                       // 0..127
    float s = 0.f;
    for (int i = 0; i < CIN; ++i) {
        float st = styles[(size_t)n * CIN + i];
        s += st * st * wsq[(size_t)o * CIN + i];
    }
    dmod[(size_t)n * COUT + o] = rsqrtf(s + 1e-8f);
}

// ---------------------------------------------------------------------------
// k2: combined FIR*convT weights per (n, parity-pair, 3x3 tap), bf16, o-major
//     Kc[n,papb,pos,o,i] = (sum_{ky,kx} fir1[my-ky]*fir1[mx-kx]*w[ky,kx,o,i])
//                          * styles[n,i] * d[n,o]
//     with my = 2*dy + 1 - pa, mx = 2*dx + 1 - pb  (fir1 = [1,3,3,1]/8)
// ---------------------------------------------------------------------------
__global__ __launch_bounds__(256) void k_combine(const float* __restrict__ w,
                                                 const float* __restrict__ styles,
                                                 const float* __restrict__ dmod,
                                                 __bf16* __restrict__ Kcb) {
    const float fir1[4] = {0.125f, 0.375f, 0.375f, 0.125f};
    int idx  = blockIdx.x;
    int o    = idx & 127;  idx >>= 7;
    int pos  = idx % 9;    idx /= 9;
    int papb = idx & 3;
    int n    = idx >> 2;
    int pa = papb >> 1, pb = papb & 1;
    int dy = pos / 3, dx = pos % 3;
    int my = 2 * dy + 1 - pa;
    int mx = 2 * dx + 1 - pb;

    float cy[3], cx[3];
#pragma unroll
    for (int k = 0; k < 3; ++k) {
        int uy = my - k;  cy[k] = (uy >= 0 && uy < 4) ? fir1[uy] : 0.f;
        int ux = mx - k;  cx[k] = (ux >= 0 && ux < 4) ? fir1[ux] : 0.f;
    }

    int i = threadIdx.x;                       // 0..255
    float acc = 0.f;
#pragma unroll
    for (int ky = 0; ky < 3; ++ky) {
#pragma unroll
        for (int kx = 0; kx < 3; ++kx) {
            float cc = cy[ky] * cx[kx];
            acc += cc * w[((size_t)(ky * 3 + kx) * COUT + o) * CIN + i];
        }
    }
    float val = acc * styles[(size_t)n * CIN + i] * dmod[(size_t)n * COUT + o];
    size_t off = ((((size_t)n * 4 + papb) * 9 + pos) * COUT + o) * CIN + i;
    Kcb[off] = (__bf16)val;
}

// ---------------------------------------------------------------------------
// TDM: fetch Kc[n,papb][pos=0..8][o=0..127][i_block of 32] -> LDS (72 KB)
// 3-D tile: X = 32 elem (2B), Y = 128 rows (stride 256 elem),
//           Z = 9 taps (stride 32768 elem = COUT*CIN).
// Per CDNA5 ISA: Y stride = tensor_dim0_stride,
//                             Z stride = tensor_dim1_stride.
// ---------------------------------------------------------------------------
__device__ __forceinline__ void tdm_load_B(const __bf16* gsrc, unsigned int lds_off) {
    unsigned long long ga = (unsigned long long)(uintptr_t)gsrc;
    v4u g0 = { 1u,                                   // count=1 (valid user D#)
               lds_off,                               // lds_addr (bytes)
               (unsigned int)(ga & 0xFFFFFFFFu),      // global_addr[31:0]
               (unsigned int)((ga >> 32) & 0x1FFFFFFu) | 0x80000000u }; // addr[56:32] | type=2
    v8i g1 = { 0x00010000,        // data_size = 1 (2 bytes); mask/flags = 0
               0x01000000,        // tensor_dim0 = 256  (bits[79:48], low 16 here)
               0x00800000,        // tensor_dim1 = 128  (bits[111:80], low 16 here)
               0x00200000,        // tile_dim0 = 32     (bits[127:112])
               0x00090080,        // tile_dim1 = 128 | tile_dim2 = 9
               256,               // tensor_dim0_stride (low 32)
               (int)0x80000000u,  // tensor_dim1_stride = 32768 (low 16 in [31:16])
               0 };
    v4i g2 = { 9, 0, 0, 0 };      // tensor_dim2 = 9
    v4i g3 = { 0, 0, 0, 0 };
#if __clang_major__ >= 23
    v8i g4 = { 0, 0, 0, 0, 0, 0, 0, 0 };
    __builtin_amdgcn_tensor_load_to_lds(g0, g1, g2, g3, g4, 0);
#else
    __builtin_amdgcn_tensor_load_to_lds(g0, g1, g2, g3, 0);
#endif
}

// ---------------------------------------------------------------------------
// k3: main fused GEMM, TDM double-buffered.
//   grid: bid = ((n*4 + papb)*64 + tile); tile -> 8x8 block of (t,s)
//   output: y[n, 2t+pa, 2s+pb, o] = sum_{dy,dx,i} Kc * x[n, t+dy-1, s+dx-1, i]
//   8 waves: wave w owns M-block (w&3) [16 pixels] and O-half (w>>2) [4 N-tiles]
// ---------------------------------------------------------------------------
__global__ __launch_bounds__(256) void k_conv_wmma(const float* __restrict__ x,
                                                   const __bf16* __restrict__ Kcb,
                                                   const float* __restrict__ bias,
                                                   float* __restrict__ out) {
    __shared__ __align__(64) __bf16 ldsA[2][100 * 32];      // 10x10 halo x 32 ch
    __shared__ __align__(64) __bf16 ldsB[2][9 * 128 * 32];  // [pos][o][i%32]

    const int tid  = threadIdx.x;
    const int lane = tid & 31;
    const int wv   = tid >> 5;
    const int mb   = wv & 3;          // M block (16 pixels)
    const int nb   = wv >> 2;         // O half (0/1)

    const int bid  = blockIdx.x;
    const int tile = bid & 63;
    const int papb = (bid >> 6) & 3;
    const int n    = bid >> 8;
    const int pa = papb >> 1, pb = papb & 1;
    const int t0 = (tile >> 3) << 3;
    const int s0 = (tile & 7) << 3;

    // per-lane fragment geometry
    const int mlocal = lane & 15;
    const int P      = mb * 16 + mlocal;
    const int dt     = P >> 3, ds = P & 7;
    const int kk0    = (lane >> 4) << 3;   // A: K half offset (0/8)
    const int kb     = (lane >> 4) << 4;   // unused (0/16); B is read at kk0 like A

    const v8f vz = {0.f, 0.f, 0.f, 0.f, 0.f, 0.f, 0.f, 0.f};
    v8f acc[4];
#pragma unroll
    for (int j = 0; j < 4; ++j) acc[j] = vz;

    const __bf16* KcBase = Kcb + (((size_t)n * 4 + papb) * 9) * COUT * CIN;
    const float*  xBase  = x + (size_t)n * H_ * W_ * CIN;

    // ---- prologue: kick TDM for ib=0, stage A halo for ib=0 ----
    if (wv == 0)
        tdm_load_B(KcBase, (unsigned int)(uintptr_t)&ldsB[0][0]);
    {
        for (int e = tid; e < 3200; e += 256) {
            int p = e >> 5, k = e & 31;
            int r = p / 10, c = p % 10;
            int gy = t0 - 1 + r, gx = s0 - 1 + c;
            float v = 0.f;
            if (gy >= 0 && gy < H_ && gx >= 0 && gx < W_)
                v = xBase[((size_t)gy * W_ + gx) * CIN + k];
            ldsA[0][e] = (__bf16)v;
        }
    }

    for (int ib = 0; ib < 8; ++ib) {           // 32-channel input blocks
        const int cur = ib & 1, nxt = cur ^ 1;

        if (ib < 7) {
            // prefetch next B via TDM (overlaps A-stage loads and this compute)
            if (wv == 0)
                tdm_load_B(KcBase + (size_t)(ib + 1) * 32,
                           (unsigned int)(uintptr_t)&ldsB[nxt][0]);
            // stage next A halo (f32 -> bf16)
            int cb = (ib + 1) * 32;
            for (int e = tid; e < 3200; e += 256) {
                int p = e >> 5, k = e & 31;
                int r = p / 10, c = p % 10;
                int gy = t0 - 1 + r, gx = s0 - 1 + c;
                float v = 0.f;
                if (gy >= 0 && gy < H_ && gx >= 0 && gx < W_)
                    v = xBase[((size_t)gy * W_ + gx) * CIN + cb + k];
                ldsA[nxt][e] = (__bf16)v;
            }
            if (wv == 0) __builtin_amdgcn_s_wait_tensorcnt(1);  // B(ib) done
        } else {
            if (wv == 0) __builtin_amdgcn_s_wait_tensorcnt(0);
        }
        __syncthreads();                        // bufA[cur]+bufB[cur] visible

        // ---- 9 taps x 4 O-tiles = 36 WMMAs, no barriers in between ----
#pragma unroll
        for (int pos = 0; pos < 9; ++pos) {
            int dy = pos / 3, dx = pos % 3;
            int pix = (dt + dy) * 10 + (ds + dx);
            const v8bf* ap = reinterpret_cast<const v8bf*>(&ldsA[cur][pix * 32 + kk0]);
            v8bf aLo = ap[0];                   // K = kk0..kk0+7
            v8bf aHi = ap[2];                   // K = 16+kk0..16+kk0+7
            v16bf afrag = __builtin_shufflevector(aLo, aHi,
                0, 1, 2, 3, 4, 5, 6, 7, 8, 9, 10, 11, 12, 13, 14, 15);
#pragma unroll
            for (int j = 0; j < 4; ++j) {
                int orow = nb * 64 + j * 16 + mlocal;
                const v8bf* bp = reinterpret_cast<const v8bf*>(
                    &ldsB[cur][pos * 4096 + orow * 32 + kk0]);
                v16bf bfrag = __builtin_shufflevector(bp[0], bp[2],
                    0, 1, 2, 3, 4, 5, 6, 7, 8, 9, 10, 11, 12, 13, 14, 15);
                acc[j] = __builtin_amdgcn_wmma_f32_16x16x32_bf16(
                    false, afrag, false, bfrag, (short)0, acc[j], false, false);
            }
        }
        __syncthreads();                        // compute done before buffers reused
    }

    // ---- epilogue: C layout lane L, vgpr r -> M = r + (L>=16)*8, N = L&15 ----
    const int mrow0 = (lane >> 4) << 3;
#pragma unroll
    for (int j = 0; j < 4; ++j) {
        int o = nb * 64 + j * 16 + mlocal;
        float bv = bias[o];
#pragma unroll
        for (int r = 0; r < 8; ++r) {
            int Pm = mb * 16 + mrow0 + r;
            int t = t0 + (Pm >> 3);
            int s = s0 + (Pm & 7);
            int arow = 2 * t + pa;
            int bcol = 2 * s + pb;
            out[(((size_t)n * 128 + arow) * 128 + bcol) * COUT + o] = acc[j][r] + bv;
        }
    }
}

// ---------------------------------------------------------------------------
extern "C" void kernel_launch(void* const* d_in, const int* in_sizes, int n_in,
                              void* d_out, int out_size, void* d_ws, size_t ws_size,
                              hipStream_t stream) {
    const float* x       = (const float*)d_in[0];   // [16,64,64,256]
    const float* latents = (const float*)d_in[1];   // [16,512]
    const float* w       = (const float*)d_in[2];   // [3,3,128,256]
    const float* style_w = (const float*)d_in[3];   // [512,256]
    const float* style_b = (const float*)d_in[4];   // [256]
    const float* b       = (const float*)d_in[5];   // [128]
    float*       out     = (float*)d_out;           // [16,128,128,128]

    char* ws = (char*)d_ws;
    float*  styles = (float*)(ws + 0);                     // 16*256*4   = 16 KB
    float*  wsq    = (float*)(ws + 16384);                 // 128*256*4  = 128 KB
    float*  dmod   = (float*)(ws + 16384 + 131072);        // 16*128*4   = 8 KB
    __bf16* Kcb    = (__bf16*)(ws + 16384 + 131072 + 8192);// 16*4*9*128*256*2 = 36 MB

    k_styles<<<N_, 256, 0, stream>>>(latents, style_w, style_b, styles);
    k_wsq<<<COUT, 256, 0, stream>>>(w, wsq);
    k_demod<<<N_, 128, 0, stream>>>(styles, wsq, dmod);
    k_combine<<<N_ * 4 * 9 * COUT, 256, 0, stream>>>(w, styles, dmod, Kcb);
    k_conv_wmma<<<N_ * 4 * 64, 256, 0, stream>>>(x, Kcb, b, out);
}
